// LSTMCRF_76038101008943
// MI455X (gfx1250) — matrix operand order repaired and checked by reading, not compiled, not run on hardware
//
#include <hip/hip_runtime.h>
#include <hip/hip_bf16.h>

// ---------------- problem constants ----------------
#define Sx 512
#define Bx 64
#define Hx 256
#define Tx 32
#define Gx 1024   // 4*H

typedef __attribute__((ext_vector_type(16))) _Float16 v16h;
typedef __attribute__((ext_vector_type(8)))  float    v8f;
typedef __attribute__((ext_vector_type(4)))  unsigned int u32x4;
typedef __attribute__((ext_vector_type(4)))  int      v4i;

#define AS1 __attribute__((address_space(1)))
#define AS3 __attribute__((address_space(3)))

#define WMMA16(a,b,c) __builtin_amdgcn_wmma_f32_16x16x32_f16(false,(a),false,(b),(short)0,(c),false,false)

#if __has_builtin(__builtin_amdgcn_global_load_async_to_lds_b128)
#define HAVE_ASYNC_LDS 1
#else
#define HAVE_ASYNC_LDS 0
#endif

static __device__ inline void async_wait_all() {
#if __has_builtin(__builtin_amdgcn_s_wait_asynccnt)
    __builtin_amdgcn_s_wait_asynccnt(0);
#else
    asm volatile("s_wait_asynccnt 0x0" ::: "memory");
#endif
}

// ---- WMMA fragment loaders (layouts per CDNA5 ISA 7.12.2) ----
// A: 16(M) x 32(K) f16, row-major source with leading dim ld (halves).
static __device__ inline v16h load_a_frag(const _Float16* p0, int ld) {
    int lane = threadIdx.x & 31;
    int m  = lane & 15;
    int bk = (lane >> 4) << 3;          // 0 or 8
    const _Float16* p = p0 + m * ld + bk;
    v16h a;
    ((u32x4*)&a)[0] = *(const u32x4*)(p);        // K = bk..bk+7
    ((u32x4*)&a)[1] = *(const u32x4*)(p + 16);   // K = bk+16..bk+23
    return a;
}
// B: 32(K) x 16(N); source is row-major W (N x K) so B[k][n] = W[n][k].
static __device__ inline v16h load_b_frag(const _Float16* p0, int ld) {
    int lane = threadIdx.x & 31;
    int n  = lane & 15;
    int bk = (lane >> 4) << 3;          // 0 or 8
    const _Float16* p = p0 + n * ld + bk;
    v16h b;
    ((u32x4*)&b)[0] = *(const u32x4*)(p);
    ((u32x4*)&b)[1] = *(const u32x4*)(p + 16);
    return b;
}

static __device__ inline float sigm(float x) { return 1.f / (1.f + __expf(-x)); }
static __device__ inline float tanh_f(float x) {
    float e = __expf(2.f * x);
    return (e - 1.f) / (e + 1.f);
}

// ---------------- prep kernels ----------------
__global__ void k_cvt_f16(const float* __restrict__ src, _Float16* __restrict__ dst, int n) {
    int i = blockIdx.x * blockDim.x + threadIdx.x;
    if (i < n) dst[i] = (_Float16)src[i];
}
__global__ void k_bias_sum(const float* a, const float* b, float* dst, int n) {
    int i = blockIdx.x * blockDim.x + threadIdx.x;
    if (i < n) dst[i] = a[i] + b[i];
}
// emb_h[(s*B+b)*H + h] = embedding[seq[b,s]][h]  (f16)
__global__ void k_embed(const int* __restrict__ seq, const float* __restrict__ emb,
                        _Float16* __restrict__ out) {
    int i  = blockIdx.x * 256 + threadIdx.x;   // < S*B*H
    int h  = i & (Hx - 1);
    int sb = i >> 8;                           // s*B + b
    int b  = sb & (Bx - 1);
    int s  = sb >> 6;
    int tok = seq[b * Sx + s];
    out[i] = (_Float16)emb[(size_t)tok * Hx + h];
}

// ---------------- input-projection GEMM: gx = emb @ Wih^T + (bih+bhh) ----------------
// M = S*B = 32768, N = 1024, K = 256. Wave -> 16x64 tile, software-pipelined K loop.
__global__ __launch_bounds__(256) void k_gx(const _Float16* __restrict__ A,
                                            const _Float16* __restrict__ W,
                                            const float* __restrict__ bsum,
                                            _Float16* __restrict__ out) {
    int wave = threadIdx.x >> 5, lane = threadIdx.x & 31;
    int m0 = blockIdx.x * 16;
    int n0 = blockIdx.y * 512 + wave * 64;
    v8f acc[4] = {};
    const _Float16* Ab = A + (size_t)m0 * Hx;
    v16h a_c  = load_a_frag(Ab, Hx);
    v16h b_c0 = load_b_frag(W + (size_t)(n0 +  0) * Hx, Hx);
    v16h b_c1 = load_b_frag(W + (size_t)(n0 + 16) * Hx, Hx);
    v16h b_c2 = load_b_frag(W + (size_t)(n0 + 32) * Hx, Hx);
    v16h b_c3 = load_b_frag(W + (size_t)(n0 + 48) * Hx, Hx);
#pragma unroll
    for (int kc = 0; kc < 8; ++kc) {
        v16h a_n, b_n0, b_n1, b_n2, b_n3;
        if (kc < 7) {                       // prefetch next K-chunk fragments
            int k1 = (kc + 1) * 32;
            a_n  = load_a_frag(Ab + k1, Hx);
            b_n0 = load_b_frag(W + (size_t)(n0 +  0) * Hx + k1, Hx);
            b_n1 = load_b_frag(W + (size_t)(n0 + 16) * Hx + k1, Hx);
            b_n2 = load_b_frag(W + (size_t)(n0 + 32) * Hx + k1, Hx);
            b_n3 = load_b_frag(W + (size_t)(n0 + 48) * Hx + k1, Hx);
        }
        acc[0] = WMMA16(a_c, b_c0, acc[0]);
        acc[1] = WMMA16(a_c, b_c1, acc[1]);
        acc[2] = WMMA16(a_c, b_c2, acc[2]);
        acc[3] = WMMA16(a_c, b_c3, acc[3]);
        if (kc < 7) { a_c = a_n; b_c0 = b_n0; b_c1 = b_n1; b_c2 = b_n2; b_c3 = b_n3; }
    }
    int n = lane & 15, mb = (lane >> 4) * 8;
#pragma unroll
    for (int nt = 0; nt < 4; ++nt) {
        int col = n0 + nt * 16 + n;
        float bv = bsum[col];
#pragma unroll
        for (int v = 0; v < 8; ++v)
            out[(size_t)(m0 + mb + v) * Gx + col] = (_Float16)(acc[nt][v] + bv);
    }
}

// ---- stage one 16x1024 f16 gx tile (32KB) into LDS: async on gfx1250 ----
static __device__ inline void stage_tile(const _Float16* __restrict__ src,
                                         _Float16* dst, int tid /*0..1023*/) {
#if HAVE_ASYNC_LDS
    __builtin_amdgcn_global_load_async_to_lds_b128(
        (AS1 v4i*)(src + (size_t)tid * 8),
        (AS3 v4i*)(dst + (size_t)tid * 8), 0, 0);
    __builtin_amdgcn_global_load_async_to_lds_b128(
        (AS1 v4i*)(src + 8192 + (size_t)tid * 8),
        (AS3 v4i*)(dst + 8192 + (size_t)tid * 8), 0, 0);
#else
    ((u32x4*)dst)[tid]        = ((const u32x4*)src)[tid];
    ((u32x4*)dst)[tid + 1024] = ((const u32x4*)src)[tid + 1024];
#endif
}

// ---------------- sequential BiLSTM ----------------
// 8 persistent workgroups of 1024 threads (32 waves): wg 0..3 forward (batch rows wg*16),
// wg 4..7 backward.  Per step: g(16x1024) = gx[t] + h(16x256) @ Whh^T ; gates -> c,h in LDS.
__global__ __launch_bounds__(1024) void k_lstm(const _Float16* __restrict__ gx_f,
                                               const _Float16* __restrict__ gx_b,
                                               const _Float16* __restrict__ WhhF,
                                               const _Float16* __restrict__ WhhB,
                                               _Float16* __restrict__ hf,
                                               _Float16* __restrict__ hb) {
    __shared__ float    g_s[16 * Gx];        // 64 KB
    __shared__ float    c_s[16 * Hx];        // 16 KB
    __shared__ _Float16 h_s[16 * Hx];        //  8 KB
    __shared__ _Float16 gxb[2][16 * Gx];     // 64 KB double buffer
    int wg  = blockIdx.x;
    int dir = wg >> 2;
    int m0  = (wg & 3) * 16;
    const _Float16* gx  = dir ? gx_b : gx_f;
    const _Float16* Whh = dir ? WhhB : WhhF;
    _Float16* hout      = dir ? hb   : hf;
    int tid = threadIdx.x;
    for (int i = tid; i < 16 * Hx; i += 1024) { c_s[i] = 0.f; h_s[i] = (_Float16)0.f; }
    // prologue: stage first tile
    int tfirst = dir ? (Sx - 1) : 0;
    stage_tile(gx + (size_t)(tfirst * Bx + m0) * Gx, gxb[0], tid);
#if HAVE_ASYNC_LDS
    async_wait_all();
#endif
    __syncthreads();
    int wave = tid >> 5, lane = tid & 31;
    int n = lane & 15, mb = (lane >> 4) * 8;

    for (int step = 0; step < Sx; ++step) {
        int t = dir ? (Sx - 1 - step) : step;
        int cur = step & 1;
        // kick off async stage of next step's tile into the other buffer
        int tn = dir ? (t - 1) : (t + 1);
        if (tn >= 0 && tn < Sx)
            stage_tile(gx + (size_t)(tn * Bx + m0) * Gx, gxb[cur ^ 1], tid);

        // GEMM: each of 32 waves owns 32 output columns (2 WMMA n-tiles)
#pragma unroll
        for (int nt = 0; nt < 2; ++nt) {
            int n0 = wave * 32 + nt * 16;
            int col = n0 + n;
            v8f acc;
#pragma unroll
            for (int v = 0; v < 8; ++v)
                acc[v] = (float)gxb[cur][(mb + v) * Gx + col];   // C init = gx + bias
            v16h b_c = load_b_frag(Whh + (size_t)n0 * Hx, Hx);
#pragma unroll
            for (int kc = 0; kc < 8; ++kc) {
                v16h a = load_a_frag(h_s + kc * 32, Hx);
                if (kc < 7) {
                    v16h b_n = load_b_frag(Whh + (size_t)n0 * Hx + (kc + 1) * 32, Hx);
                    acc = WMMA16(a, b_c, acc);
                    b_c = b_n;
                } else {
                    acc = WMMA16(a, b_c, acc);
                }
            }
#pragma unroll
            for (int v = 0; v < 8; ++v) g_s[(mb + v) * Gx + col] = acc[v];
        }
        __syncthreads();
        // gates: i,f,g,o = g[:, 0:256, 256:512, 512:768, 768:1024]
        for (int idx = tid; idx < 16 * Hx; idx += 1024) {
            int r = idx >> 8, j = idx & 255;
            float ig = g_s[r * Gx + j];
            float fg = g_s[r * Gx + 256 + j];
            float gg = g_s[r * Gx + 512 + j];
            float og = g_s[r * Gx + 768 + j];
            float c  = sigm(fg) * c_s[idx] + sigm(ig) * tanh_f(gg);
            float h  = sigm(og) * tanh_f(c);
            c_s[idx] = c;
            _Float16 hh = (_Float16)h;
            h_s[idx] = hh;
            hout[(size_t)(t * Bx + m0 + r) * Hx + j] = hh;
        }
#if HAVE_ASYNC_LDS
        async_wait_all();     // staged tile complete before releasing the barrier
#endif
        __syncthreads();
    }
}

// ---------------- FC: logits = [hf,hb] @ fcW^T + fcb  (M=32768, N=32, K=512) ----------------
__global__ __launch_bounds__(256) void k_fc(const _Float16* __restrict__ hf,
                                            const _Float16* __restrict__ hb,
                                            const _Float16* __restrict__ fcWh,
                                            const float* __restrict__ fcb,
                                            float* __restrict__ logits) {
    int wave = threadIdx.x >> 5, lane = threadIdx.x & 31;
    int m0 = (blockIdx.x * 8 + wave) * 16;
    v8f acc[2] = {};
    for (int kc = 0; kc < 16; ++kc) {
        int k0 = kc * 32;
        const _Float16* A = (k0 < Hx) ? (hf + (size_t)m0 * Hx + k0)
                                      : (hb + (size_t)m0 * Hx + (k0 - Hx));
        v16h a = load_a_frag(A, Hx);
#pragma unroll
        for (int nt = 0; nt < 2; ++nt) {
            v16h bb = load_b_frag(fcWh + (size_t)(nt * 16) * (2 * Hx) + k0, 2 * Hx);
            acc[nt] = WMMA16(a, bb, acc[nt]);
        }
    }
    int n = lane & 15, mb = (lane >> 4) * 8;
#pragma unroll
    for (int nt = 0; nt < 2; ++nt) {
        int col = nt * 16 + n;
        float bv = fcb[col];
#pragma unroll
        for (int v = 0; v < 8; ++v)
            logits[(size_t)(m0 + mb + v) * Tx + col] = acc[nt][v] + bv;
    }
}

__global__ void k_zero(float* p) { if (threadIdx.x == 0) p[0] = 0.f; }

// ---------------- CRF negative log-likelihood (one wave32 per batch row) ----------------
__global__ __launch_bounds__(32) void k_crf(const float* __restrict__ logits,
                                            const int* __restrict__ mask,
                                            const int* __restrict__ labels,
                                            const float* __restrict__ start,
                                            const float* __restrict__ endt,
                                            const float* __restrict__ trans,
                                            float* __restrict__ loss) {
    __shared__ float trans_s[Tx * Tx];
    __shared__ float score[Tx];
    __shared__ float red[Tx];
    int b = blockIdx.x, t = threadIdx.x;
    for (int i = t; i < Tx * Tx; i += Tx) trans_s[i] = trans[i];
    score[t] = start[t] + logits[(size_t)b * Tx + t];
    __syncthreads();

    float num = 0.f;
    if (t == 0) {
        int tag0 = labels[b * Sx];
        num = start[tag0] + logits[(size_t)b * Tx + tag0];
        int prev = tag0;
        int cnt = mask[b * Sx];
        for (int s = 1; s < Sx; ++s) {
            int tg = labels[b * Sx + s];
            int m  = mask[b * Sx + s];
            if (m) num += trans_s[prev * Tx + tg] + logits[(size_t)(s * Bx + b) * Tx + tg];
            prev = tg;
            cnt += m;
        }
        if (cnt < 1) cnt = 1;
        num += endt[labels[b * Sx + (cnt - 1)]];
        red[0] = num;  // stash
    }
    __syncthreads();
    num = red[0];

    for (int s = 1; s < Sx; ++s) {
        float em = logits[(size_t)(s * Bx + b) * Tx + t];
        int m = mask[b * Sx + s];
        float mx = -1e30f;
        for (int k = 0; k < Tx; ++k) mx = fmaxf(mx, score[k] + trans_s[k * Tx + t]);
        float sum = 0.f;
        for (int k = 0; k < Tx; ++k) sum += __expf(score[k] + trans_s[k * Tx + t] - mx);
        float nxt = mx + __logf(sum) + em;
        __syncthreads();
        if (m) score[t] = nxt;
        __syncthreads();
    }
    red[t] = score[t] + endt[t];
    __syncthreads();
    if (t == 0) {
        float mx = -1e30f;
        for (int k = 0; k < Tx; ++k) mx = fmaxf(mx, red[k]);
        float sum = 0.f;
        for (int k = 0; k < Tx; ++k) sum += __expf(red[k] - mx);
        float den = mx + __logf(sum);
        atomicAdd(loss, den - num);   // loss = -sum(num - den)
    }
}

// ---------------- Viterbi (one wave32 per batch row) + backtrack ----------------
__global__ __launch_bounds__(32) void k_vit(const float* __restrict__ logits,
                                            const int* __restrict__ mask,
                                            const float* __restrict__ start,
                                            const float* __restrict__ endt,
                                            const float* __restrict__ trans,
                                            unsigned char* __restrict__ bp,
                                            float* __restrict__ preds) {
    __shared__ float trans_s[Tx * Tx];
    __shared__ float score[Tx];
    __shared__ float red[Tx];
    int b = blockIdx.x, t = threadIdx.x;
    for (int i = t; i < Tx * Tx; i += Tx) trans_s[i] = trans[i];
    score[t] = start[t] + logits[(size_t)b * Tx + t];
    __syncthreads();
    for (int s = 1; s < Sx; ++s) {
        float em = logits[(size_t)(s * Bx + b) * Tx + t];
        int m = mask[b * Sx + s];
        float best = -1e30f; int bi = 0;
        for (int k = 0; k < Tx; ++k) {
            float v = score[k] + trans_s[k * Tx + t];
            if (v > best) { best = v; bi = k; }
        }
        float nxt = best + em;
        bp[(size_t)(s * Bx + b) * Tx + t] = (unsigned char)(m ? bi : t);
        __syncthreads();
        if (m) score[t] = nxt;
        __syncthreads();
    }
    red[t] = score[t] + endt[t];
    __syncthreads();
    if (t == 0) {
        float best = red[0]; int last = 0;
        for (int k = 1; k < Tx; ++k) if (red[k] > best) { best = red[k]; last = k; }
        int tag = last;
        preds[b * Sx + (Sx - 1)] = (float)last;
        for (int s = Sx - 2; s >= 0; --s) {
            tag = bp[(size_t)((s + 1) * Bx + b) * Tx + tag];
            preds[b * Sx + s] = (float)tag;
        }
    }
}

// ---------------- launch ----------------
extern "C" void kernel_launch(void* const* d_in, const int* in_sizes, int n_in,
                              void* d_out, int out_size, void* d_ws, size_t ws_size,
                              hipStream_t stream) {
    const int*   seq   = (const int*)  d_in[0];
    const int*   mask  = (const int*)  d_in[1];
    const int*   lab   = (const int*)  d_in[2];
    const float* emb   = (const float*)d_in[3];
    const float* WihF  = (const float*)d_in[4];
    const float* WhhF  = (const float*)d_in[5];
    const float* bihF  = (const float*)d_in[6];
    const float* bhhF  = (const float*)d_in[7];
    const float* WihB  = (const float*)d_in[8];
    const float* WhhB  = (const float*)d_in[9];
    const float* bihB  = (const float*)d_in[10];
    const float* bhhB  = (const float*)d_in[11];
    const float* fcW   = (const float*)d_in[12];
    const float* fcb   = (const float*)d_in[13];
    const float* stt   = (const float*)d_in[14];
    const float* endt  = (const float*)d_in[15];
    const float* trans = (const float*)d_in[16];

    char* ws = (char*)d_ws;
    const size_t SZ_W = (size_t)Gx * Hx * sizeof(_Float16);            // 512 KB
    _Float16* wihf_h = (_Float16*)(ws + 0 * SZ_W);
    _Float16* whhf_h = (_Float16*)(ws + 1 * SZ_W);
    _Float16* wihb_h = (_Float16*)(ws + 2 * SZ_W);
    _Float16* whhb_h = (_Float16*)(ws + 3 * SZ_W);
    size_t off = 4 * SZ_W;
    _Float16* fcw_h  = (_Float16*)(ws + off); off += (size_t)Tx * 2 * Hx * 2;   // 32 KB
    float*    bsum_f = (float*)   (ws + off); off += Gx * 4;
    float*    bsum_b = (float*)   (ws + off); off += Gx * 4;
    _Float16* emb_h  = (_Float16*)(ws + off); off += (size_t)Sx * Bx * Hx * 2;  // 16 MB
    _Float16* gx_f   = (_Float16*)(ws + off); off += (size_t)Sx * Bx * Gx * 2;  // 64 MB
    _Float16* gx_b   = (_Float16*)(ws + off); off += (size_t)Sx * Bx * Gx * 2;  // 64 MB
    _Float16* hf_h   = (_Float16*)(ws + off); off += (size_t)Sx * Bx * Hx * 2;  // 16 MB
    _Float16* hb_h   = (_Float16*)(ws + off); off += (size_t)Sx * Bx * Hx * 2;  // 16 MB
    float*    logits = (float*)   (ws + off); off += (size_t)Sx * Bx * Tx * 4;  //  4 MB
    unsigned char* bp = (unsigned char*)(ws + off);                             //  1 MB

    float* preds = (float*)d_out;
    float* lossp = ((float*)d_out) + (size_t)Bx * Sx;

    const int NW = Gx * Hx; // 262144
    k_cvt_f16<<<(NW + 255) / 256, 256, 0, stream>>>(WihF, wihf_h, NW);
    k_cvt_f16<<<(NW + 255) / 256, 256, 0, stream>>>(WhhF, whhf_h, NW);
    k_cvt_f16<<<(NW + 255) / 256, 256, 0, stream>>>(WihB, wihb_h, NW);
    k_cvt_f16<<<(NW + 255) / 256, 256, 0, stream>>>(WhhB, whhb_h, NW);
    k_cvt_f16<<<(Tx * 2 * Hx + 255) / 256, 256, 0, stream>>>(fcW, fcw_h, Tx * 2 * Hx);
    k_bias_sum<<<4, 256, 0, stream>>>(bihF, bhhF, bsum_f, Gx);
    k_bias_sum<<<4, 256, 0, stream>>>(bihB, bhhB, bsum_b, Gx);

    k_embed<<<(Sx * Bx * Hx) / 256, 256, 0, stream>>>(seq, emb, emb_h);

    dim3 ggrid(Sx * Bx / 16, 2);
    k_gx<<<ggrid, 256, 0, stream>>>(emb_h, wihf_h, bsum_f, gx_f);
    k_gx<<<ggrid, 256, 0, stream>>>(emb_h, wihb_h, bsum_b, gx_b);

    k_lstm<<<8, 1024, 0, stream>>>(gx_f, gx_b, whhf_h, whhb_h, hf_h, hb_h);

    k_fc<<<Sx * Bx / 16 / 8, 256, 0, stream>>>(hf_h, hb_h, fcw_h, fcb, logits);

    k_zero<<<1, 32, 0, stream>>>(lossp);
    k_crf<<<Bx, 32, 0, stream>>>(logits, mask, lab, stt, endt, trans, lossp);
    k_vit<<<Bx, 32, 0, stream>>>(logits, mask, stt, endt, trans, bp, preds);
}
